// Encoder_14448269983845
// MI455X (gfx1250) — matrix operand order repaired and checked by reading, not compiled, not run on hardware
//
#include <hip/hip_runtime.h>
#include <cstdint>

// ---------------------------------------------------------------------------
// Encoder: attention (Q/K/V + softmax + AV) followed by an LSTM over T steps.
// B=512, T=128, N=256, H=256, 4H=1024.  All GEMMs use bf16 WMMA
// (v_wmma_f32_16x16x32_bf16) with fp32 accumulate.  Each wave computes a
// 16x64 (or 16x32) output strip; A fragments are software-pipelined so the
// next k-step's loads overlap the current WMMAs.  Weight staging into LDS
// uses the Tensor Data Mover when available.
// ---------------------------------------------------------------------------

#define B_  512
#define T_  128
#define N_  256
#define H_  256
#define G4_ 1024

typedef __attribute__((ext_vector_type(16))) __bf16       v16bf;
typedef __attribute__((ext_vector_type(8)))  float        v8f;
typedef __attribute__((ext_vector_type(4)))  float        f32x4;
typedef __attribute__((ext_vector_type(4)))  unsigned int u32x4;
typedef __attribute__((ext_vector_type(8)))  int          i32x8;
typedef __attribute__((ext_vector_type(4)))  int          i32x4;

#if defined(__has_builtin)
#if __has_builtin(__builtin_amdgcn_tensor_load_to_lds)
#define HAVE_TDM 1
#endif
#endif
#ifndef HAVE_TDM
#define HAVE_TDM 0
#endif

// ---- WMMA fragment loaders (wave32, 16x16x32 bf16) -------------------------
// A (16x32, MxK): lane L: hf=L/16, m=L%16.  VGPRs 0-3 hold K=hf*8..+7,
// VGPRs 4-7 hold K=16+hf*8..+7  -> two contiguous 16B loads per lane.
__device__ __forceinline__ v16bf load_a_frag(const __bf16* base, int ld) {
    const int lane = threadIdx.x & 31;
    const int hf   = lane >> 4;
    const int m    = lane & 15;
    const __bf16* r = base + (size_t)m * ld;
    union { v16bf v; f32x4 q[2]; } u;
    u.q[0] = *reinterpret_cast<const f32x4*>(r + hf * 8);
    u.q[1] = *reinterpret_cast<const f32x4*>(r + 16 + hf * 8);
    return u.v;
}

// B (32x16, KxN) supplied as Bt row-major with Bt[n][k] = B[k][n].
__device__ __forceinline__ v16bf load_bt_frag(const __bf16* base, int ld) {
    const int lane = threadIdx.x & 31;
    const int hf   = lane >> 4;
    const int n    = lane & 15;
    const __bf16* c = base + (size_t)n * ld + hf * 8;
    union { v16bf v; f32x4 q[2]; } u;
    u.q[0] = *reinterpret_cast<const f32x4*>(c);
    u.q[1] = *reinterpret_cast<const f32x4*>(c + 16);
    return u.v;
}

__device__ __forceinline__ v8f wmma_bf16(const v16bf& a, const v16bf& b, const v8f& c) {
    return __builtin_amdgcn_wmma_f32_16x16x32_bf16(false, a, false, b,
                                                   (short)0, c, false, false);
}

// 16 x (CT*16) strip with software-pipelined A fragments: the A load for
// k-step ks+1 is issued before the WMMAs of ks, hiding global latency.
template <int CT, int KTILES>
__device__ __forceinline__ void gemm_strip(const __bf16* Abase, int lda,
                                           const __bf16* Btbase, int ldb,
                                           v8f (&acc)[CT]) {
    v16bf a_cur = load_a_frag(Abase, lda);
#pragma unroll
    for (int ks = 0; ks < KTILES; ++ks) {
        v16bf a_nxt = a_cur;
        if (ks + 1 < KTILES)
            a_nxt = load_a_frag(Abase + (ks + 1) * 32, lda);
#pragma unroll
        for (int c = 0; c < CT; ++c) {
            const v16bf b = load_bt_frag(Btbase + (size_t)(c * 16) * ldb + ks * 32, ldb);
            acc[c] = wmma_bf16(a_cur, b, acc[c]);
        }
        a_cur = a_nxt;
    }
}

// ---- Tensor Data Mover: 1-D copy of `elems` bf16 elements into LDS ---------
#if HAVE_TDM
__device__ __forceinline__ void tdm_load_1d(unsigned int lds_off, const void* gsrc,
                                            unsigned int elems) {
    const unsigned long long ga = (unsigned long long)(uintptr_t)gsrc;
    u32x4 g0 = { 1u,                                    // count=1, user descriptor
                 lds_off,                               // lds_addr (bytes)
                 (unsigned int)ga,                      // global_addr[31:0]
                 (unsigned int)((ga >> 32) & 0x1FFFFFFu) | (2u << 30) }; // [56:32]|type=2
    i32x8 g1 = { (int)(1u << 16),                       // data_size=1 (2 bytes), mask=0
                 (int)((elems & 0xFFFFu) << 16),        // tensor_dim0[15:0]
                 (int)((elems >> 16) & 0xFFFFu),        // tensor_dim0[31:16]
                 (int)((elems & 0xFFFFu) << 16),        // tile_dim0
                 0,
                 (int)elems,                            // tensor_dim0_stride lo
                 0, 0 };
    i32x4 z4 = {};
#if defined(__clang_major__) && (__clang_major__ >= 23)
    i32x8 z8 = {};
    __builtin_amdgcn_tensor_load_to_lds(g0, g1, z4, z4, z8, 0);
#else
    __builtin_amdgcn_tensor_load_to_lds(g0, g1, z4, z4, 0);
#endif
}
#endif

__device__ __forceinline__ float sigmoidf_(float x) { return 1.0f / (1.0f + __expf(-x)); }

// ===========================================================================
// Prep kernels
// ===========================================================================
__global__ void cvt_bf16_kernel(const float* __restrict__ src, __bf16* __restrict__ dst, int n) {
    int i = blockIdx.x * blockDim.x + threadIdx.x;
    if (i < n) dst[i] = (__bf16)src[i];
}

// Xp[b][n][t] = X[b][t][n]  (fp32 -> bf16)
__global__ void xpose_kernel(const float* __restrict__ x, __bf16* __restrict__ xp) {
    long i = (long)blockIdx.x * blockDim.x + threadIdx.x;   // B*N*T
    long t = i & (T_ - 1);
    long n = (i >> 7) & (N_ - 1);
    long b = i >> 15;
    xp[i] = (__bf16)x[(b * T_ + t) * N_ + n];
}

__global__ void init_state_kernel(float* __restrict__ c, __bf16* __restrict__ hbf) {
    int i = blockIdx.x * blockDim.x + threadIdx.x;          // B*H
    c[i]   = 0.0f;
    hbf[i] = (__bf16)0.0f;
}

// ===========================================================================
// QKV:  Out[b] (N x T) = Xp[b] @ W^T.  W tile staged in LDS via TDM.
// sel 0->Q, 1->K stored (N x T); sel 2->V stored transposed (T x N).
// ===========================================================================
__global__ __launch_bounds__(256) void qkv_kernel(
        const __bf16* __restrict__ xp,
        const __bf16* __restrict__ wq, const __bf16* __restrict__ wk,
        const __bf16* __restrict__ wv,
        __bf16* __restrict__ qb, __bf16* __restrict__ kb, __bf16* __restrict__ vt) {
    __shared__ __align__(16) __bf16 wlds[T_ * T_];          // 32 KB
    const int sel = blockIdx.x % 3;
    const int b   = blockIdx.x / 3;
    const __bf16* w = (sel == 0) ? wq : (sel == 1) ? wk : wv;

#if HAVE_TDM
    if (threadIdx.x < 32) {                                 // one TDM op per WG
        tdm_load_1d((unsigned int)(uintptr_t)(&wlds[0]), w, T_ * T_);
        __builtin_amdgcn_s_wait_tensorcnt(0);
    }
#else
    {
        const f32x4* s = reinterpret_cast<const f32x4*>(w);
        f32x4* d = reinterpret_cast<f32x4*>(wlds);
        for (int i = threadIdx.x; i < (T_ * T_) / 8; i += 256) d[i] = s[i];
    }
#endif
    __syncthreads();

    const __bf16* a_b  = xp + (size_t)b * N_ * T_;
    const int wave = threadIdx.x >> 5;
    const int lane = threadIdx.x & 31;
    const int hf = lane >> 4, n = lane & 15;

    // 16x64 strips: 16 row-tiles x 2 col-strips = 32 strips
    for (int s = wave; s < 32; s += 8) {
        const int m0 = (s >> 1) * 16;                       // N dim
        const int j0 = (s & 1) * 64;                        // T dim
        v8f acc[4] = {};
        gemm_strip<4, T_ / 32>(a_b + (size_t)m0 * T_, T_, wlds + (size_t)j0 * T_, T_, acc);
        if (sel == 2) {
            __bf16* o = vt + (size_t)b * T_ * N_;
#pragma unroll
            for (int c = 0; c < 4; ++c)
#pragma unroll
                for (int r = 0; r < 8; ++r)
                    o[(size_t)(j0 + c * 16 + n) * N_ + (m0 + r + 8 * hf)] = (__bf16)acc[c][r];
        } else {
            __bf16* o = ((sel == 0) ? qb : kb) + (size_t)b * N_ * T_;
#pragma unroll
            for (int c = 0; c < 4; ++c)
#pragma unroll
                for (int r = 0; r < 8; ++r)
                    o[(size_t)(m0 + r + 8 * hf) * T_ + (j0 + c * 16 + n)] = (__bf16)acc[c][r];
        }
    }
}

// ===========================================================================
// Attention: scores = Q K^T / 16, softmax rows, Xa = A @ V.
// One workgroup per (batch, 32-row block).
// ===========================================================================
__global__ __launch_bounds__(256) void attn_kernel(
        const __bf16* __restrict__ qb, const __bf16* __restrict__ kb,
        const __bf16* __restrict__ vt, __bf16* __restrict__ xa) {
    __shared__ __align__(16) float  slds[32 * N_];          // 32 KB
    __shared__ __align__(16) __bf16 abf [32 * N_];          // 16 KB
    __shared__ float red[256];                              // 1 KB
    const int b  = blockIdx.x >> 3;
    const int i0 = (blockIdx.x & 7) * 32;
    const __bf16* q_b = qb + (size_t)b * N_ * T_;
    const __bf16* k_b = kb + (size_t)b * N_ * T_;
    const __bf16* v_b = vt + (size_t)b * T_ * N_;
    const int wave = threadIdx.x >> 5;
    const int lane = threadIdx.x & 31;
    const int hf = lane >> 4, n = lane & 15;

    // --- scores (32 x 256): 16x64 strips, 8 strips -> 1 per wave ---
    {
        const int mt = wave >> 2;                           // 0..1
        const int j0 = (wave & 3) * 64;
        v8f acc[4] = {};
        gemm_strip<4, T_ / 32>(q_b + (size_t)(i0 + mt * 16) * T_, T_,
                               k_b + (size_t)j0 * T_, T_, acc);
#pragma unroll
        for (int c = 0; c < 4; ++c)
#pragma unroll
            for (int r = 0; r < 8; ++r)
                slds[(mt * 16 + r + 8 * hf) * N_ + j0 + c * 16 + n] =
                    acc[c][r] * 0.0625f;                    // 1/sqrt(N)
    }
    __syncthreads();

    // --- softmax: 8 threads per row (32 cols each) ---
    {
        const int r   = threadIdx.x >> 3;
        const int sub = threadIdx.x & 7;
        float* row = slds + r * N_;
        float m = -3.4e38f;
        for (int j = sub * 32; j < sub * 32 + 32; ++j) m = fmaxf(m, row[j]);
        red[threadIdx.x] = m;
        __syncthreads();
        if (sub == 0) {
            float mm = red[r * 8];
            for (int q = 1; q < 8; ++q) mm = fmaxf(mm, red[r * 8 + q]);
            red[r * 8] = mm;
        }
        __syncthreads();
        const float mm = red[r * 8];
        float s = 0.0f;
        for (int j = sub * 32; j < sub * 32 + 32; ++j) {
            const float e = __expf(row[j] - mm);
            row[j] = e; s += e;
        }
        __syncthreads();                                    // red[r*8] done being read
        red[threadIdx.x] = s;
        __syncthreads();
        if (sub == 0) {
            float ss = 0.0f;
            for (int q = 0; q < 8; ++q) ss += red[r * 8 + q];
            red[r * 8] = 1.0f / ss;
        }
        __syncthreads();
        const float inv = red[r * 8];
        __bf16* arow = abf + r * N_;
        for (int j = sub * 32; j < sub * 32 + 32; ++j) arow[j] = (__bf16)(row[j] * inv);
    }
    __syncthreads();

    // --- Xa (32 x 128) = A @ V : 16x32 strips, 8 strips -> 1 per wave ---
    {
        const int mt = wave >> 2;                           // 0..1
        const int t0 = (wave & 3) * 32;
        v8f acc[2] = {};
        gemm_strip<2, N_ / 32>(abf + (size_t)(mt * 16) * N_, N_,
                               v_b + (size_t)t0 * N_, N_, acc);
        __bf16* o = xa + (size_t)b * T_ * N_;
#pragma unroll
        for (int c = 0; c < 2; ++c)
#pragma unroll
            for (int r = 0; r < 8; ++r)   // write transposed: Xa[b][t][n_row]
                o[(size_t)(t0 + c * 16 + n) * N_ + (i0 + mt * 16 + r + 8 * hf)] =
                    (__bf16)acc[c][r];
    }
}

// ===========================================================================
// Gx = Xa @ W_ih^T for ALL timesteps:  (B*T x 4H), K=N.  Stored (T, B, 4H).
// Each wave: 16x64 strip.  65536 strips -> 8192 WGs.
// ===========================================================================
__global__ __launch_bounds__(256) void gx_kernel(
        const __bf16* __restrict__ xa, const __bf16* __restrict__ wih,
        float* __restrict__ gx) {
    const long w  = (long)blockIdx.x * 8 + (threadIdx.x >> 5);
    const long rt = w >> 4;                                 // 4096 row tiles
    const int  j0 = (int)(w & 15) * 64;                     // 16 col strips
    const long m0 = rt * 16;
    v8f acc[4] = {};
    gemm_strip<4, N_ / 32>(xa + m0 * N_, N_, wih + (size_t)j0 * N_, N_, acc);
    const int lane = threadIdx.x & 31;
    const int hf = lane >> 4, n = lane & 15;
#pragma unroll
    for (int c = 0; c < 4; ++c)
#pragma unroll
        for (int r = 0; r < 8; ++r) {
            const long m  = m0 + r + 8 * hf;
            const long bb = m >> 7;                         // / T
            const long tt = m & (T_ - 1);
            gx[(tt * B_ + bb) * G4_ + j0 + c * 16 + n] = acc[c][r];
        }
}

// ===========================================================================
// Recurrent step:  G = Gx[t] + h_bf @ W_hh^T   (512 x 1024, K=256).
// 512 strips of 16x64 -> 64 WGs; accumulator seeded from Gx (C operand).
// ===========================================================================
__global__ __launch_bounds__(256) void lstm_gemm_kernel(
        const __bf16* __restrict__ hbf, const __bf16* __restrict__ whh,
        const float* __restrict__ gx, float* __restrict__ gbuf, int t) {
    const int w  = blockIdx.x * 8 + (threadIdx.x >> 5);     // 0..511
    const int m0 = (w >> 4) * 16;                           // batch rows
    const int j0 = (w & 15) * 64;
    const int lane = threadIdx.x & 31;
    const int hf = lane >> 4, n = lane & 15;
    const float* g = gx + (size_t)t * B_ * G4_;
    v8f acc[4];
#pragma unroll
    for (int c = 0; c < 4; ++c)
#pragma unroll
        for (int r = 0; r < 8; ++r)
            acc[c][r] = g[(size_t)(m0 + r + 8 * hf) * G4_ + j0 + c * 16 + n];
    gemm_strip<4, H_ / 32>(hbf + (size_t)m0 * H_, H_, whh + (size_t)j0 * H_, H_, acc);
#pragma unroll
    for (int c = 0; c < 4; ++c)
#pragma unroll
        for (int r = 0; r < 8; ++r)
            gbuf[(size_t)(m0 + r + 8 * hf) * G4_ + j0 + c * 16 + n] = acc[c][r];
}

__global__ void lstm_point_kernel(
        const float* __restrict__ gbuf,
        const float* __restrict__ bih, const float* __restrict__ bhh,
        float* __restrict__ c, __bf16* __restrict__ hbf,
        float* __restrict__ out, int t) {
    const int idx = blockIdx.x * blockDim.x + threadIdx.x;  // B*H
    const int b = idx >> 8;
    const int h = idx & (H_ - 1);
    const float* g = gbuf + (size_t)b * G4_;
    const float gi = g[h]          + bih[h]          + bhh[h];
    const float gf = g[H_ + h]     + bih[H_ + h]     + bhh[H_ + h];
    const float gg = g[2 * H_ + h] + bih[2 * H_ + h] + bhh[2 * H_ + h];
    const float go = g[3 * H_ + h] + bih[3 * H_ + h] + bhh[3 * H_ + h];
    const float iv = sigmoidf_(gi);
    const float fv = sigmoidf_(gf);
    const float gv = tanhf(gg);
    const float ov = sigmoidf_(go);
    const float cv = fv * c[idx] + iv * gv;
    c[idx] = cv;
    const float hv = ov * tanhf(cv);
    out[((size_t)b * T_ + t) * H_ + h] = hv;
    hbf[idx] = (__bf16)hv;
}

// ===========================================================================
// Host launcher
// ===========================================================================
extern "C" void kernel_launch(void* const* d_in, const int* in_sizes, int n_in,
                              void* d_out, int out_size, void* d_ws, size_t ws_size,
                              hipStream_t stream) {
    (void)in_sizes; (void)n_in; (void)out_size; (void)ws_size;
    const float* X   = (const float*)d_in[0];
    const float* Wq  = (const float*)d_in[1];
    const float* Wk  = (const float*)d_in[2];
    const float* Wv  = (const float*)d_in[3];
    const float* Wih = (const float*)d_in[4];
    const float* Whh = (const float*)d_in[5];
    const float* bih = (const float*)d_in[6];
    const float* bhh = (const float*)d_in[7];
    float* out = (float*)d_out;

    char* p = (char*)d_ws;
    auto alloc = [&](size_t bytes) -> void* {
        void* r = p; p += (bytes + 255) & ~(size_t)255; return r;
    };
    __bf16* wq_bf  = (__bf16*)alloc((size_t)T_ * T_ * 2);
    __bf16* wk_bf  = (__bf16*)alloc((size_t)T_ * T_ * 2);
    __bf16* wv_bf  = (__bf16*)alloc((size_t)T_ * T_ * 2);
    __bf16* wih_bf = (__bf16*)alloc((size_t)G4_ * N_ * 2);
    __bf16* whh_bf = (__bf16*)alloc((size_t)G4_ * H_ * 2);
    __bf16* xp     = (__bf16*)alloc((size_t)B_ * N_ * T_ * 2);
    __bf16* qb     = (__bf16*)alloc((size_t)B_ * N_ * T_ * 2);
    __bf16* kb     = (__bf16*)alloc((size_t)B_ * N_ * T_ * 2);
    __bf16* vt     = (__bf16*)alloc((size_t)B_ * T_ * N_ * 2);
    __bf16* xa     = (__bf16*)alloc((size_t)B_ * T_ * N_ * 2);
    float*  gx     = (float* )alloc((size_t)B_ * T_ * G4_ * 4);
    float*  gbuf   = (float* )alloc((size_t)B_ * G4_ * 4);
    float*  cst    = (float* )alloc((size_t)B_ * H_ * 4);
    __bf16* hbf    = (__bf16*)alloc((size_t)B_ * H_ * 2);

    cvt_bf16_kernel<<<(T_ * T_ + 255) / 256, 256, 0, stream>>>(Wq, wq_bf, T_ * T_);
    cvt_bf16_kernel<<<(T_ * T_ + 255) / 256, 256, 0, stream>>>(Wk, wk_bf, T_ * T_);
    cvt_bf16_kernel<<<(T_ * T_ + 255) / 256, 256, 0, stream>>>(Wv, wv_bf, T_ * T_);
    cvt_bf16_kernel<<<(G4_ * N_ + 255) / 256, 256, 0, stream>>>(Wih, wih_bf, G4_ * N_);
    cvt_bf16_kernel<<<(G4_ * H_ + 255) / 256, 256, 0, stream>>>(Whh, whh_bf, G4_ * H_);

    xpose_kernel<<<(B_ * N_ * T_) / 256, 256, 0, stream>>>(X, xp);

    qkv_kernel<<<B_ * 3, 256, 0, stream>>>(xp, wq_bf, wk_bf, wv_bf, qb, kb, vt);
    attn_kernel<<<B_ * 8, 256, 0, stream>>>(qb, kb, vt, xa);
    gx_kernel<<<(B_ * T_ / 16) * (G4_ / 64) / 8, 256, 0, stream>>>(xa, wih_bf, gx);

    init_state_kernel<<<(B_ * H_) / 256, 256, 0, stream>>>(cst, hbf);
    for (int t = 0; t < T_; ++t) {
        lstm_gemm_kernel<<<(B_ / 16) * (G4_ / 64) / 8, 256, 0, stream>>>(
            hbf, whh_bf, gx, gbuf, t);
        lstm_point_kernel<<<(B_ * H_) / 256, 256, 0, stream>>>(
            gbuf, bih, bhh, cst, hbf, out, t);
    }
}
